// DGCNN_cls_85160611545541
// MI455X (gfx1250) — matrix operand order repaired and checked by reading, not compiled, not run on hardware
//
#include <hip/hip_runtime.h>
#include <hip/hip_bf16.h>

// ---------------------------------------------------------------------------
// DGCNN-cls forward for MI455X (gfx1250, wave32).
// GEMM-shaped ops -> v_wmma_f32_16x16x32_f16 (f32 accumulate), operands
// pre-packed to f16 with K padded to a multiple of 32 so the inner loop is
// pure b128 loads + wmma. KNN top-k + per-point kmeans -> VALU + LDS, with
// the kmeans feature gather done via GLOBAL_LOAD_ASYNC_TO_LDS (ASYNCcnt).
// ---------------------------------------------------------------------------

typedef __attribute__((ext_vector_type(16))) _Float16 v16h;
typedef __attribute__((ext_vector_type(8)))  _Float16 v8h;
typedef __attribute__((ext_vector_type(8)))  float    v8f;

#define NPTS   2048
#define BATCH  8
#define KMAX   40
#define KCLUS  15
#define KMIT   10

// ---------------------------------------------------------------------------
// Squared norms per point: sqn[b][n] = sum_c h[b][c][n]^2
// ---------------------------------------------------------------------------
__global__ void sqnorm_kernel(const float* __restrict__ h, float* __restrict__ sqn,
                              int C, int N_) {
  int t = blockIdx.x * blockDim.x + threadIdx.x;
  if (t >= BATCH * N_) return;
  int b = t / N_, m = t % N_;
  const float* hb = h + (long)b * C * N_;
  float s = 0.f;
  for (int c = 0; c < C; ++c) { float v = hb[c * N_ + m]; s += v * v; }
  sqn[t] = s;
}

// ---------------------------------------------------------------------------
// KNN: for each query point keep the 40 nearest (ranking key = |m|^2 - 2<q,m>,
// equivalent to full squared distance). Sorted ascending -> matches top_k(-d).
// ---------------------------------------------------------------------------
template <int C>
__global__ void knn_topk_kernel(const float* __restrict__ h,
                                const float* __restrict__ sqn,
                                int* __restrict__ idx, int N_) {
  int t = blockIdx.x * blockDim.x + threadIdx.x;
  if (t >= BATCH * N_) return;
  int b = t / N_, n = t % N_;
  const float* hb = h + (long)b * C * N_;

  float xq[C];
#pragma unroll
  for (int c = 0; c < C; ++c) xq[c] = hb[c * N_ + n];

  float bkey[KMAX];
  int   bidx[KMAX];
#pragma unroll
  for (int j = 0; j < KMAX; ++j) { bkey[j] = 3.0e38f; bidx[j] = 0; }

  const float* sq = sqn + (long)b * N_;
  for (int m = 0; m < N_; ++m) {
    float dot = 0.f;
#pragma unroll
    for (int c = 0; c < C; ++c) dot += xq[c] * hb[c * N_ + m];
    float key = sq[m] - 2.0f * dot;
    if (key < bkey[KMAX - 1]) {
      int j = KMAX - 1;
      while (j > 0 && bkey[j - 1] > key) {
        bkey[j] = bkey[j - 1]; bidx[j] = bidx[j - 1]; --j;
      }
      bkey[j] = key; bidx[j] = m;
    }
  }
  int* op = idx + (long)t * KMAX;
  for (int j = 0; j < KMAX; ++j) op[j] = bidx[j];
}

// ---------------------------------------------------------------------------
// Per-point kmeans (40 pts, 15 clusters, 10 iters) + cluster means.
// One wave (32 lanes) per group, everything in LDS. The random-index feature
// gather goes global->LDS directly via GLOBAL_LOAD_ASYNC_TO_LDS_B32
// (ASYNCcnt), no VGPR round trip.
// Writes nb TRANSPOSED in f16: per batch (ncols x Kpad), ncols = 15*N,
// column j = k*N + n, zero-padded rows C..Kpad-1. This makes every GEMM
// B-fragment 16 contiguous halves (2 x b128 per lane).
// ---------------------------------------------------------------------------
template <int C, int KP>
__global__ void kmeans_nb_kernel(const float* __restrict__ h,
                                 const int* __restrict__ idx,
                                 _Float16* __restrict__ nbT, int N_) {
  __shared__ float feat[KMAX * C];
  __shared__ float cent[KCLUS * C];
  __shared__ float sums[KCLUS * C];
  __shared__ float cnt[KCLUS];
  __shared__ int   lab[KMAX];

  int g = blockIdx.x;           // g = b*N + n
  int b = g / N_, n = g % N_;
  int lane = threadIdx.x;       // blockDim.x == 32
  const float* hb = h + (long)b * C * N_;
  const int* ip = idx + (long)g * KMAX;

  // async gather: LDS[feat[t]] = h[b][c][idx[j]]
  for (int t = lane; t < KMAX * C; t += 32) {
    int j = t / C, c = t % C;
    unsigned lds_addr = (unsigned)(size_t)(&feat[t]);   // low dword = AS3 offset
    unsigned long long gaddr = (unsigned long long)(size_t)(hb + c * N_ + ip[j]);
    asm volatile("global_load_async_to_lds_b32 %0, %1, off"
                 :: "v"(lds_addr), "v"(gaddr) : "memory");
  }
  asm volatile("s_wait_asynccnt 0x0" ::: "memory");
  __syncthreads();
  for (int t = lane; t < KCLUS * C; t += 32) cent[t] = feat[t];  // first 15 pts
  __syncthreads();

  for (int it = 0; it <= KMIT; ++it) {
    // assignment
    for (int j = lane; j < KMAX; j += 32) {
      float bd = 3.0e38f; int bk = 0;
      for (int k = 0; k < KCLUS; ++k) {
        float d = 0.f;
        for (int c = 0; c < C; ++c) {
          float df = feat[j * C + c] - cent[k * C + c];
          d += df * df;
        }
        if (d < bd) { bd = d; bk = k; }
      }
      lab[j] = bk;
    }
    __syncthreads();
    // counts + sums (no atomics: each lane owns (k,c) slots)
    for (int k = lane; k < KCLUS; k += 32) {
      float cc = 0.f;
      for (int j = 0; j < KMAX; ++j) cc += (lab[j] == k) ? 1.0f : 0.0f;
      cnt[k] = cc;
    }
    for (int t = lane; t < KCLUS * C; t += 32) {
      int k = t / C, c = t % C;
      float s = 0.f;
      for (int j = 0; j < KMAX; ++j) if (lab[j] == k) s += feat[j * C + c];
      sums[t] = s;
    }
    __syncthreads();
    if (it < KMIT) {   // centroid update (empty cluster keeps old centroid)
      for (int t = lane; t < KCLUS * C; t += 32) {
        int k = t / C;
        if (cnt[k] > 0.f) cent[t] = sums[t] / cnt[k];
      }
      __syncthreads();
    }
  }
  // final means: sums / max(cnt, 1)  (empty cluster -> 0); write f16 transposed
  for (int t = lane; t < KCLUS * KP; t += 32) {
    int k = t / KP, c = t % KP;
    float mv = (c < C) ? sums[k * C + c] / fmaxf(cnt[k], 1.0f) : 0.0f;
    nbT[((long)b * KCLUS * N_ + (long)k * N_ + n) * KP + c] = (_Float16)mv;
  }
}

// ---------------------------------------------------------------------------
// Pack weights (O x K f32, row-major) -> f16 (O x Kpad), zero-padded.
// ---------------------------------------------------------------------------
__global__ void pack_w_f16(const float* __restrict__ W, _Float16* __restrict__ Wh,
                           int O, int K, int Kpad) {
  int t = blockIdx.x * blockDim.x + threadIdx.x;
  if (t >= O * Kpad) return;
  int o = t / Kpad, k = t % Kpad;
  Wh[t] = (k < K) ? (_Float16)W[(long)o * K + k] : (_Float16)0.0f;
}

// Pack activations (B, C, N) f32 -> transposed f16 (B, N, Kpad).
__global__ void pack_bT_f16(const float* __restrict__ H, _Float16* __restrict__ Bh,
                            int C, int N_, int Kpad) {
  long t = (long)blockIdx.x * blockDim.x + threadIdx.x;
  long total = (long)BATCH * N_ * Kpad;
  if (t >= total) return;
  int k = (int)(t % Kpad);
  long r = t / Kpad;
  int n = (int)(r % N_);
  int b = (int)(r / N_);
  Bh[t] = (k < C) ? (_Float16)H[((long)b * C + k) * N_ + n] : (_Float16)0.0f;
}

// ---------------------------------------------------------------------------
// WMMA GEMM: Y[b] (M x ncols, f32) = A (M x Kpad, f16, shared) x B[b]
// (Kpad x ncols, stored TRANSPOSED as ncols x Kpad f16).
// One wave computes a 16x64 output tile: one A fragment feeds 4 wmma per
// K-step. All fragment loads are contiguous b128 pairs (both operands use
// k = 16*(lane/16) + s: A lane = row, B lane = col, 16 contiguous K each).
// ---------------------------------------------------------------------------
__global__ __launch_bounds__(32)
void wmma_gemm_kernel(const _Float16* __restrict__ A,
                      const _Float16* __restrict__ Bt,
                      float* __restrict__ Y,
                      int M, int Kpad, int ncols,
                      long strideB, long strideY) {
  const int lane = threadIdx.x & 31;
  const int colBase = blockIdx.x * 64;
  const int rowTile = blockIdx.y * 16;
  const int b = blockIdx.z;
  const _Float16* Bp = Bt + (long)b * strideB;
  float* Yp = Y + (long)b * strideY;

  const int m  = lane & 15;          // A row / B,D column within tile
  const int kh = lane >> 4;          // A: K half select; D: row-offset select
  const int kb = kh * 16;            // B: K block select

  const _Float16* Ap = A + (long)(rowTile + m) * Kpad;
  const _Float16* B0 = Bp + (long)(colBase + m) * Kpad + kb;
  const long cstr = (long)16 * Kpad; // 16 columns ahead

  v8f acc0 = {}, acc1 = {}, acc2 = {}, acc3 = {};
  for (int k0 = 0; k0 < Kpad; k0 += 32) {
    v8h alo = *(const v8h*)(Ap + k0 + kh * 16);
    v8h ahi = *(const v8h*)(Ap + k0 + kh * 16 + 8);
    v16h a;
#pragma unroll
    for (int e = 0; e < 8; ++e) { a[e] = alo[e]; a[8 + e] = ahi[e]; }
    v16h b0 = *(const v16h*)(B0 + k0);
    v16h b1 = *(const v16h*)(B0 + cstr + k0);
    v16h b2 = *(const v16h*)(B0 + 2 * cstr + k0);
    v16h b3 = *(const v16h*)(B0 + 3 * cstr + k0);
    acc0 = __builtin_amdgcn_wmma_f32_16x16x32_f16(false, a, false, b0, (short)0, acc0, false, false);
    acc1 = __builtin_amdgcn_wmma_f32_16x16x32_f16(false, a, false, b1, (short)0, acc1, false, false);
    acc2 = __builtin_amdgcn_wmma_f32_16x16x32_f16(false, a, false, b2, (short)0, acc2, false, false);
    acc3 = __builtin_amdgcn_wmma_f32_16x16x32_f16(false, a, false, b3, (short)0, acc3, false, false);
  }
#pragma unroll
  for (int r = 0; r < 8; ++r) {
    int row = rowTile + r + 8 * kh;
    float* yr = Yp + (long)row * ncols + colBase + m;
    yr[0]  = acc0[r];
    yr[16] = acc1[r];
    yr[32] = acc2[r];
    yr[48] = acc3[r];
  }
}

// ---------------------------------------------------------------------------
// BN stats per channel over (B, ncols): writes fused scale/shift.
// ---------------------------------------------------------------------------
__global__ void bn_stats_kernel(const float* __restrict__ y, int M, long ncols,
                                const float* __restrict__ g,
                                const float* __restrict__ beta,
                                float* __restrict__ scale,
                                float* __restrict__ shift) {
  __shared__ float ss[256], ss2[256];
  int o = blockIdx.x;
  float s = 0.f, s2 = 0.f;
  for (int b = 0; b < BATCH; ++b) {
    const float* yp = y + ((long)b * M + o) * ncols;
    for (long j = threadIdx.x; j < ncols; j += blockDim.x) {
      float v = yp[j]; s += v; s2 += v * v;
    }
  }
  ss[threadIdx.x] = s; ss2[threadIdx.x] = s2;
  __syncthreads();
  for (int st = 128; st > 0; st >>= 1) {
    if ((int)threadIdx.x < st) {
      ss[threadIdx.x]  += ss[threadIdx.x + st];
      ss2[threadIdx.x] += ss2[threadIdx.x + st];
    }
    __syncthreads();
  }
  if (threadIdx.x == 0) {
    float tot  = (float)(BATCH * (double)ncols);
    float mean = ss[0] / tot;
    float var  = ss2[0] / tot - mean * mean;
    float inv  = rsqrtf(var + 1e-5f);
    float sc   = g[o] * inv;
    scale[o] = sc;
    shift[o] = beta[o] - mean * sc;
  }
}

// out[b][o][n] = max_m relu(y[b][o][m*N+n]*scale+shift)
__global__ void bn_relu_maxm_kernel(const float* __restrict__ y,
                                    const float* __restrict__ scale,
                                    const float* __restrict__ shift,
                                    float* __restrict__ out, int M, int N_) {
  long t = (long)blockIdx.x * blockDim.x + threadIdx.x;
  long total = (long)BATCH * M * N_;
  if (t >= total) return;
  int n = (int)(t % N_);
  long r = t / N_;
  int o = (int)(r % M);
  int b = (int)(r / M);
  const float* yp = y + ((long)b * M + o) * (KCLUS * (long)N_) + n;
  float sc = scale[o], sh = shift[o];
  float mx = 0.f;   // relu >= 0 so 0 is a valid identity for max-of-relu
  for (int m = 0; m < KCLUS; ++m) {
    float v = yp[(long)m * N_] * sc + sh;
    mx = fmaxf(mx, fmaxf(v, 0.f));
  }
  out[((long)b * M + o) * N_ + n] = mx;
}

// out[b][o] = max_n relu(y[b][o][n]*scale+shift)
__global__ void bn_relu_maxn_kernel(const float* __restrict__ y,
                                    const float* __restrict__ scale,
                                    const float* __restrict__ shift,
                                    float* __restrict__ out, int M, int N_) {
  __shared__ float sm[256];
  int o = blockIdx.x % M;
  int b = blockIdx.x / M;
  const float* yp = y + ((long)b * M + o) * N_;
  float sc = scale[o], sh = shift[o];
  float mx = 0.f;
  for (int n = threadIdx.x; n < N_; n += blockDim.x)
    mx = fmaxf(mx, fmaxf(yp[n] * sc + sh, 0.f));
  sm[threadIdx.x] = mx;
  __syncthreads();
  for (int st = 128; st > 0; st >>= 1) {
    if ((int)threadIdx.x < st) sm[threadIdx.x] = fmaxf(sm[threadIdx.x], sm[threadIdx.x + st]);
    __syncthreads();
  }
  if (threadIdx.x == 0) out[(long)b * M + o] = sm[0];
}

// y[b][o] = dot(W[o], h[b]) + bias[o]   (optionally relu)
__global__ void fc_kernel(const float* __restrict__ h, const float* __restrict__ W,
                          const float* __restrict__ bias, float* __restrict__ y,
                          int O, int K, int do_relu) {
  int t = blockIdx.x * blockDim.x + threadIdx.x;
  if (t >= BATCH * O) return;
  int b = t / O, o = t % O;
  float s = bias ? bias[o] : 0.f;
  const float* hp = h + (long)b * K;
  const float* wp = W + (long)o * K;
  for (int k = 0; k < K; ++k) s += hp[k] * wp[k];
  y[t] = do_relu ? fmaxf(s, 0.f) : s;
}

// BN over the batch dim (8 samples) + relu
__global__ void bn_fc_relu_kernel(const float* __restrict__ y,
                                  const float* __restrict__ g,
                                  const float* __restrict__ beta,
                                  float* __restrict__ out, int O) {
  int o = blockIdx.x * blockDim.x + threadIdx.x;
  if (o >= O) return;
  float s = 0.f, s2 = 0.f;
  for (int b = 0; b < BATCH; ++b) { float v = y[b * O + o]; s += v; s2 += v * v; }
  float mean = s / BATCH;
  float var  = s2 / BATCH - mean * mean;
  float inv  = rsqrtf(var + 1e-5f);
  float sc = g[o] * inv, sh = beta[o] - mean * sc;
  for (int b = 0; b < BATCH; ++b)
    out[b * O + o] = fmaxf(y[b * O + o] * sc + sh, 0.f);
}

// ---------------------------------------------------------------------------
// Host orchestration
// ---------------------------------------------------------------------------
static inline size_t align256(size_t x) { return (x + 255) & ~(size_t)255; }

extern "C" void kernel_launch(void* const* d_in, const int* in_sizes, int n_in,
                              void* d_out, int out_size, void* d_ws, size_t ws_size,
                              hipStream_t stream) {
  const int N_ = NPTS;
  const float* x = (const float*)d_in[0];
  const float* Wc[4] = {(const float*)d_in[1], (const float*)d_in[4],
                        (const float*)d_in[7], (const float*)d_in[10]};
  const float* gc[4] = {(const float*)d_in[2], (const float*)d_in[5],
                        (const float*)d_in[8], (const float*)d_in[11]};
  const float* bc[4] = {(const float*)d_in[3], (const float*)d_in[6],
                        (const float*)d_in[9], (const float*)d_in[12]};
  const float* W5 = (const float*)d_in[13];
  const float* g5 = (const float*)d_in[14];
  const float* b5 = (const float*)d_in[15];
  const float* W6 = (const float*)d_in[16];
  const float* bb6 = (const float*)d_in[17];
  const float* g6 = (const float*)d_in[18];
  const float* b6 = (const float*)d_in[19];
  const float* W7 = (const float*)d_in[20];
  const float* bb7 = (const float*)d_in[21];
  const float* g7 = (const float*)d_in[22];
  const float* b7 = (const float*)d_in[23];
  const float* W8 = (const float*)d_in[24];
  const float* bb8 = (const float*)d_in[25];

  char* wp = (char*)d_ws;
  size_t off = 0;
  auto alloc = [&](size_t bytes) -> void* {
    void* p = wp + off;
    off = align256(off + bytes);
    return p;
  };
  float*     sqn   = (float*)alloc((size_t)BATCH * N_ * 4);
  int*       idx   = (int*)  alloc((size_t)BATCH * N_ * KMAX * 4);
  float*     hA    = (float*)alloc((size_t)BATCH * 64 * N_ * 4);
  float*     hB    = (float*)alloc((size_t)BATCH * 64 * N_ * 4);
  _Float16*  nbT   = (_Float16*)alloc((size_t)BATCH * KCLUS * N_ * 64 * 2);
  _Float16*  hT    = (_Float16*)alloc((size_t)BATCH * N_ * 64 * 2);
  _Float16*  Wh    = (_Float16*)alloc((size_t)1024 * 64 * 2);
  float*     ybuf  = (float*)alloc((size_t)BATCH * 1024 * N_ * 4);  // conv y & y5
  float*     scale = (float*)alloc(1024 * 4);
  float*     shift = (float*)alloc(1024 * 4);
  float*     hmax  = (float*)alloc((size_t)BATCH * 1024 * 4);
  float*     fcy   = (float*)alloc((size_t)BATCH * 512 * 4);
  float*     fca   = (float*)alloc((size_t)BATCH * 512 * 4);
  (void)ws_size; (void)in_sizes; (void)n_in; (void)out_size;

  const int P = BATCH * N_;               // 16384 points total
  const int ncols = KCLUS * N_;           // 30720 GEMM columns per batch

  const float* hin = x;
  float* hout = hA;
  int C = 3;
  for (int L = 0; L < 4; ++L) {
    const int Kpad = (C == 3) ? 32 : 64;
    sqnorm_kernel<<<(P + 255) / 256, 256, 0, stream>>>(hin, sqn, C, N_);
    if (C == 3) {
      knn_topk_kernel<3><<<(P + 63) / 64, 64, 0, stream>>>(hin, sqn, idx, N_);
      kmeans_nb_kernel<3, 32><<<P, 32, 0, stream>>>(hin, idx, nbT, N_);
    } else {
      knn_topk_kernel<64><<<(P + 63) / 64, 64, 0, stream>>>(hin, sqn, idx, N_);
      kmeans_nb_kernel<64, 64><<<P, 32, 0, stream>>>(hin, idx, nbT, N_);
    }
    pack_w_f16<<<(64 * Kpad + 255) / 256, 256, 0, stream>>>(Wc[L], Wh, 64, C, Kpad);
    wmma_gemm_kernel<<<dim3(ncols / 64, 64 / 16, BATCH), 32, 0, stream>>>(
        Wh, nbT, ybuf, 64, Kpad, ncols, (long)ncols * Kpad, (long)64 * ncols);
    bn_stats_kernel<<<64, 256, 0, stream>>>(ybuf, 64, (long)ncols, gc[L], bc[L],
                                            scale, shift);
    bn_relu_maxm_kernel<<<((long)BATCH * 64 * N_ + 255) / 256, 256, 0, stream>>>(
        ybuf, scale, shift, hout, 64, N_);
    hin = hout;
    hout = (hout == hA) ? hB : hA;
    C = 64;
  }

  // layer 5: 64 -> 1024 pointwise conv, BN over (B,N), relu, max over N
  pack_w_f16<<<(1024 * 64 + 255) / 256, 256, 0, stream>>>(W5, Wh, 1024, 64, 64);
  pack_bT_f16<<<((long)BATCH * N_ * 64 + 255) / 256, 256, 0, stream>>>(hin, hT, 64, N_, 64);
  wmma_gemm_kernel<<<dim3(N_ / 64, 1024 / 16, BATCH), 32, 0, stream>>>(
      Wh, hT, ybuf, 1024, 64, N_, (long)N_ * 64, (long)1024 * N_);
  bn_stats_kernel<<<1024, 256, 0, stream>>>(ybuf, 1024, (long)N_, g5, b5, scale, shift);
  bn_relu_maxn_kernel<<<BATCH * 1024, 256, 0, stream>>>(ybuf, scale, shift, hmax,
                                                        1024, N_);

  // FC head: 1024 -> 512 -> 128 -> 40
  fc_kernel<<<(BATCH * 512 + 255) / 256, 256, 0, stream>>>(hmax, W6, bb6, fcy,
                                                           512, 1024, 0);
  bn_fc_relu_kernel<<<(512 + 63) / 64, 64, 0, stream>>>(fcy, g6, b6, fca, 512);
  fc_kernel<<<(BATCH * 128 + 255) / 256, 256, 0, stream>>>(fca, W7, bb7, fcy,
                                                           128, 512, 0);
  bn_fc_relu_kernel<<<(128 + 63) / 64, 64, 0, stream>>>(fcy, g7, b7, fca, 128);
  fc_kernel<<<(BATCH * 40 + 63) / 64, 64, 0, stream>>>(fca, W8, bb8,
                                                       (float*)d_out, 40, 128, 1);
}
